// MultiSimilarity_1219770712681
// MI455X (gfx1250) — matrix operand order repaired and checked by reading, not compiled, not run on hardware
//
#include <hip/hip_runtime.h>
#include <cstdint>

typedef __attribute__((ext_vector_type(16))) _Float16 v16h;
typedef __attribute__((ext_vector_type(8)))  float    v8f;
typedef uint32_t u32x4 __attribute__((ext_vector_type(4)));

#define NROW 8192
#define DDIM 192
#define DW   96          // dwords per f16 row (192 halves)
#define LDSS 100         // LDS row stride in dwords (96 + 4 pad: 16B-aligned, conflict-free)
#define JBATCH 64        // columns staged per barrier round
#define JSPLIT 4
#define NBATCH (NROW / JBATCH)          // 128 total batches
#define BPER   (NBATCH / JSPLIT)        // 32 batches per y-split
#define ONE_EPS (1.0f - 1e-5f)
#define MARGIN 0.1f

union ABFrag { u32x4 q[2]; v16h h; };

// A-matrix 16x32 f16 lane layout: lane ln holds row M=ln (both halves);
// half hh=lane>>4 selects K offset: K = 16*hh + e (contiguous, matches B)
__device__ __forceinline__ v16h load_a_frag(const uint32_t* __restrict__ rowp, int kc, int hh) {
  ABFrag ua;
  const uint32_t* p = rowp + kc * 16 + hh * 8;   // dwords
  ua.q[0] = *(const u32x4*)(p);
  ua.q[1] = *(const u32x4*)(p + 4);
  return ua.h;
}

__global__ __launch_bounds__(256)
void ms_cvt(const float* __restrict__ f, uint32_t* __restrict__ o, int n) {
  int i = blockIdx.x * blockDim.x + threadIdx.x;
  if (i < n) {
    union { _Float16 h[2]; uint32_t u; } pk;
    pk.h[0] = (_Float16)f[2 * i];
    pk.h[1] = (_Float16)f[2 * i + 1];
    o[i] = pk.u;
  }
}

// Expand int64 labels -> per-row int32 (lab[i] = labels[i/2], n_utt = 2)
__global__ __launch_bounds__(256)
void ms_lab(const long long* __restrict__ labels, int* __restrict__ lab32) {
  int i = blockIdx.x * blockDim.x + threadIdx.x;
  if (i < NROW) lab32[i] = (int)labels[i >> 1];
}

// ---------------- Pass 1: per-row sumsq, min over same-label, max over diff-label ----------
__global__ __launch_bounds__(128)
void ms_pass1(const uint32_t* __restrict__ f16u, const int* __restrict__ lab32,
              float* __restrict__ ssP, float* __restrict__ mnP, float* __restrict__ mxP)
{
  __shared__ __align__(16) uint32_t lds[JBATCH * LDSS];   // 25.6 KB
  __shared__ int ldsLab[JBATCH];
  const int tid  = threadIdx.x;
  const int wave = tid >> 5, lane = tid & 31;
  const int hh = lane >> 4, ln = lane & 15;
  const int rowStrip = blockIdx.x * 64 + wave * 16;
  const int sc = tid >> 1;              // staging col 0..63
  const int sh = (tid & 1) * 48;        // staging dword-half offset

  v16h afr[6];
  {
    const uint32_t* rowp = f16u + (size_t)(rowStrip + ln) * DW;
#pragma unroll
    for (int kc = 0; kc < 6; ++kc) afr[kc] = load_a_frag(rowp, kc, hh);
  }
  int labr[8];
#pragma unroll
  for (int r = 0; r < 8; ++r) labr[r] = lab32[rowStrip + 8 * hh + r];

  float ss[8], mn[8], mx[8];
#pragma unroll
  for (int r = 0; r < 8; ++r) { ss[r] = 0.f; mn[r] = __builtin_inff(); mx[r] = -__builtin_inff(); }

  const int blo = blockIdx.y * BPER, bhi = blo + BPER;
  for (int jb = blo; jb < bhi; ++jb) {
    const int jbase64 = jb * JBATCH;
    __syncthreads();
    {
      const uint32_t* gp = f16u + (size_t)(jbase64 + sc) * DW + sh;
      uint32_t* lp = &lds[sc * LDSS + sh];
#pragma unroll
      for (int q = 0; q < 12; ++q)
        *(u32x4*)(lp + 4 * q) = *(const u32x4*)(gp + 4 * q);
      if (tid < JBATCH) ldsLab[tid] = lab32[jbase64 + tid];
    }
    __syncthreads();

#pragma unroll
    for (int t = 0; t < 4; ++t) {
      const int labj = ldsLab[t * 16 + ln];
      v8f acc = {};
#pragma unroll
      for (int kc = 0; kc < 6; ++kc) {
        ABFrag ub;   // B 32x16: N = ln, K = e + 16*hh within chunk
        const uint32_t* bp = &lds[(t * 16 + ln) * LDSS + kc * 16 + hh * 8];
        ub.q[0] = *(const u32x4*)(bp);
        ub.q[1] = *(const u32x4*)(bp + 4);
        acc = __builtin_amdgcn_wmma_f32_16x16x32_f16(false, afr[kc], false, ub.h,
                                                     (short)0, acc, false, false);
      }
#pragma unroll
      for (int r = 0; r < 8; ++r) {
        float s = acc[r];
        ss[r] += s * s;
        if (labj == labr[r]) mn[r] = fminf(mn[r], s);
        else                 mx[r] = fmaxf(mx[r], s);
      }
    }
  }
#pragma unroll
  for (int off = 1; off < 16; off <<= 1) {
#pragma unroll
    for (int r = 0; r < 8; ++r) {
      ss[r] += __shfl_xor(ss[r], off, 16);
      mn[r] = fminf(mn[r], __shfl_xor(mn[r], off, 16));
      mx[r] = fmaxf(mx[r], __shfl_xor(mx[r], off, 16));
    }
  }
  if (ln == 0) {
    const int base = blockIdx.y * NROW;
#pragma unroll
    for (int r = 0; r < 8; ++r) {
      int row = rowStrip + 8 * hh + r;
      ssP[base + row] = ss[r]; mnP[base + row] = mn[r]; mxP[base + row] = mx[r];
    }
  }
}

// ---------------- Combine pass-1 partials -> per-row inv-norm + thresholds -----------------
__global__ __launch_bounds__(256)
void ms_combine1(const float* __restrict__ ssP, const float* __restrict__ mnP,
                 const float* __restrict__ mxP, float* __restrict__ inva,
                 float* __restrict__ tneg, float* __restrict__ tpos)
{
  int row = blockIdx.x * blockDim.x + threadIdx.x;
  if (row >= NROW) return;
  float ss = 0.f, mn = __builtin_inff(), mx = -__builtin_inff();
#pragma unroll
  for (int y = 0; y < JSPLIT; ++y) {
    ss += ssP[y * NROW + row];
    mn = fminf(mn, mnP[y * NROW + row]);
    mx = fmaxf(mx, mxP[y * NROW + row]);
  }
  float inv = 1.f / fmaxf(sqrtf(ss), 1e-12f);
  float mp = mn * inv;                                    // min over same-label of S
  float minposS = (mp < ONE_EPS) ? mp : __builtin_inff(); // empty-set semantics
  inva[row] = inv;
  tneg[row] = minposS - MARGIN;    // keep neg iff S >  tneg  (inf -> never)
  tpos[row] = mx * inv + MARGIN;   // keep pos iff S <  tpos  (-inf -> never)
}

// ---------------- Pass 2: masked exp-sums + any-flags ----------------
__global__ __launch_bounds__(128)
void ms_pass2(const uint32_t* __restrict__ f16u, const int* __restrict__ lab32,
              const float* __restrict__ inva, const float* __restrict__ tneg,
              const float* __restrict__ tpos,
              float* __restrict__ psP, float* __restrict__ nsP, uint32_t* __restrict__ flP)
{
  __shared__ __align__(16) uint32_t lds[JBATCH * LDSS];
  __shared__ int ldsLab[JBATCH];
  const int tid  = threadIdx.x;
  const int wave = tid >> 5, lane = tid & 31;
  const int hh = lane >> 4, ln = lane & 15;
  const int rowStrip = blockIdx.x * 64 + wave * 16;
  const int sc = tid >> 1;
  const int sh = (tid & 1) * 48;

  v16h afr[6];
  {
    const uint32_t* rowp = f16u + (size_t)(rowStrip + ln) * DW;
#pragma unroll
    for (int kc = 0; kc < 6; ++kc) afr[kc] = load_a_frag(rowp, kc, hh);
  }
  int labr[8]; float iv[8], tn[8], tp[8], ps[8], ns[8];
#pragma unroll
  for (int r = 0; r < 8; ++r) {
    int row = rowStrip + 8 * hh + r;
    labr[r] = lab32[row];
    iv[r] = inva[row]; tn[r] = tneg[row]; tp[r] = tpos[row];
    ps[r] = 0.f; ns[r] = 0.f;
  }
  unsigned hp = 0u, hn = 0u;

  const int blo = blockIdx.y * BPER, bhi = blo + BPER;
  for (int jb = blo; jb < bhi; ++jb) {
    const int jbase64 = jb * JBATCH;
    __syncthreads();
    {
      const uint32_t* gp = f16u + (size_t)(jbase64 + sc) * DW + sh;
      uint32_t* lp = &lds[sc * LDSS + sh];
#pragma unroll
      for (int q = 0; q < 12; ++q)
        *(u32x4*)(lp + 4 * q) = *(const u32x4*)(gp + 4 * q);
      if (tid < JBATCH) ldsLab[tid] = lab32[jbase64 + tid];
    }
    __syncthreads();

#pragma unroll
    for (int t = 0; t < 4; ++t) {
      const int labj = ldsLab[t * 16 + ln];
      v8f acc = {};
#pragma unroll
      for (int kc = 0; kc < 6; ++kc) {
        ABFrag ub;
        const uint32_t* bp = &lds[(t * 16 + ln) * LDSS + kc * 16 + hh * 8];
        ub.q[0] = *(const u32x4*)(bp);
        ub.q[1] = *(const u32x4*)(bp + 4);
        acc = __builtin_amdgcn_wmma_f32_16x16x32_f16(false, afr[kc], false, ub.h,
                                                     (short)0, acc, false, false);
      }
#pragma unroll
      for (int r = 0; r < 8; ++r) {
        float s = acc[r] * iv[r];
        if (labj == labr[r]) {
          if (s < ONE_EPS && s < tp[r]) { ps[r] += __expf(-2.f * (s - 0.5f)); hp |= 1u << r; }
        } else {
          if (s > tn[r])                { ns[r] += __expf(50.f * (s - 0.5f)); hn |= 1u << r; }
        }
      }
    }
  }
#pragma unroll
  for (int off = 1; off < 16; off <<= 1) {
#pragma unroll
    for (int r = 0; r < 8; ++r) {
      ps[r] += __shfl_xor(ps[r], off, 16);
      ns[r] += __shfl_xor(ns[r], off, 16);
    }
    hp |= (unsigned)__shfl_xor((int)hp, off, 16);
    hn |= (unsigned)__shfl_xor((int)hn, off, 16);
  }
  if (ln == 0) {
    const int base = blockIdx.y * NROW;
#pragma unroll
    for (int r = 0; r < 8; ++r) {
      int row = rowStrip + 8 * hh + r;
      psP[base + row] = ps[r];
      nsP[base + row] = ns[r];
      flP[base + row] = ((hp >> r) & 1u) | (((hn >> r) & 1u) << 1);
    }
  }
}

// ---------------- Final scalar reduction ----------------
__global__ __launch_bounds__(256)
void ms_final(const float* __restrict__ psP, const float* __restrict__ nsP,
              const uint32_t* __restrict__ flP, float* __restrict__ out)
{
  __shared__ float sl[256], sk[256];
  int tid = threadIdx.x;
  float lsum = 0.f, ksum = 0.f;
  for (int row = tid; row < NROW; row += 256) {
    float ps = 0.f, ns = 0.f; uint32_t fl = 0u;
#pragma unroll
    for (int y = 0; y < JSPLIT; ++y) {
      ps += psP[y * NROW + row]; ns += nsP[y * NROW + row]; fl |= flP[y * NROW + row];
    }
    bool hasp = (fl & 1u) != 0u, hasn = (fl & 2u) != 0u;
    if (hasp && hasn) lsum += 0.5f * log1pf(ps) + 0.02f * log1pf(ns);
    if (!hasn) ksum += 1.f;
  }
  sl[tid] = lsum; sk[tid] = ksum; __syncthreads();
  for (int s = 128; s > 0; s >>= 1) {
    if (tid < s) { sl[tid] += sl[tid + s]; sk[tid] += sk[tid + s]; }
    __syncthreads();
  }
  if (tid == 0) { out[0] = sl[0] / (float)NROW; out[1] = sk[0] * 100.f / (float)NROW; }
}

extern "C" void kernel_launch(void* const* d_in, const int* in_sizes, int n_in,
                              void* d_out, int out_size, void* d_ws, size_t ws_size,
                              hipStream_t stream) {
  const float*     feats  = (const float*)d_in[0];      // [4096,2,192] f32
  const long long* labels = (const long long*)d_in[1];  // [4096] int64
  float* out = (float*)d_out;                           // [loss, prec1]

  // Workspace carve-up (dword units)
  uint32_t* f16u = (uint32_t*)d_ws;                     // 8192*96 dwords of packed f16
  float* wsf = (float*)d_ws;
  size_t o = (size_t)NROW * DW;
  int*   lab32 = (int*)(wsf + o); o += NROW;
  float* ssP  = wsf + o; o += (size_t)JSPLIT * NROW;
  float* mnP  = wsf + o; o += (size_t)JSPLIT * NROW;
  float* mxP  = wsf + o; o += (size_t)JSPLIT * NROW;
  float* inva = wsf + o; o += NROW;
  float* tneg = wsf + o; o += NROW;
  float* tpos = wsf + o; o += NROW;
  float* psP  = wsf + o; o += (size_t)JSPLIT * NROW;
  float* nsP  = wsf + o; o += (size_t)JSPLIT * NROW;
  uint32_t* flP = (uint32_t*)(wsf + o);

  const int ndw = NROW * DW;
  ms_cvt<<<(ndw + 255) / 256, 256, 0, stream>>>(feats, f16u, ndw);
  ms_lab<<<NROW / 256, 256, 0, stream>>>(labels, lab32);

  dim3 grid(NROW / 64, JSPLIT);
  ms_pass1<<<grid, 128, 0, stream>>>(f16u, lab32, ssP, mnP, mxP);
  ms_combine1<<<NROW / 256, 256, 0, stream>>>(ssP, mnP, mxP, inva, tneg, tpos);
  ms_pass2<<<grid, 128, 0, stream>>>(f16u, lab32, inva, tneg, tpos, psP, nsP, flP);
  ms_final<<<1, 256, 0, stream>>>(psP, nsP, flP, out);
}
